// MultiHeadCoAttention_88192858456334
// MI455X (gfx1250) — matrix operand order repaired and checked by reading, not compiled, not run on hardware
//
#include <hip/hip_runtime.h>
#include <math.h>

// MI455X / gfx1250 implementation of MultiHeadCoAttention.
// bf16 WMMA (16x16x32, f32 accum); flash-attention fused softmax.
// Ping-pong double-buffered fragment loads (no register rotation copies),
// pointer-bump addressing with immediate offsets.
// Workspace requirement: 88 MiB.

typedef __attribute__((ext_vector_type(16))) __bf16 v16bf;
typedef __attribute__((ext_vector_type(4)))  __bf16 v4bf;
typedef __attribute__((ext_vector_type(8)))  float  v8f;

#define DI __device__ __forceinline__

constexpr int BATCH = 4;
constexpr int NCH   = 2;
constexpr int T     = 1024;
constexpr int U     = 1024;   // n_units
constexpr int H     = 8;
constexpr int DK    = 128;
constexpr int DCO   = 256;    // nch * d_k
constexpr int MROWS = BATCH * NCH * T; // 8192
constexpr int NK    = U / 32;          // 32 k-steps per GEMM

union FragU { v16bf v; uint4 q[2]; };

// A-operand (16x32 bf16): lane holds row M=lane%16.
// Caller pre-adds (lane>=16 ? 8 : 0); chunks at +0 and +16.
DI v16bf load_a_at(const __bf16* p) {
  FragU f;
  f.q[0] = *(const uint4*)(p);
  f.q[1] = *(const uint4*)(p + 16);
  return f.v;
}

// B-operand (32x16 bf16): lane holds column N=lane%16.
// Caller pre-adds (lane>=16 ? 8 : 0); chunks at +0 and +16.
DI v16bf load_b_at(const __bf16* p) {
  FragU f;
  f.q[0] = *(const uint4*)(p);
  f.q[1] = *(const uint4*)(p + 16);
  return f.v;
}

DI v8f wmma_bf16(v16bf a, v16bf b, v8f c) {
  return __builtin_amdgcn_wmma_f32_16x16x32_bf16(
      false, a, false, b, (short)0, c, false, false);
}

// ---------------------------------------------------------------------------
// Kernel 0: f32 -> bf16 convert (vectorized x4)
// ---------------------------------------------------------------------------
__global__ void cvt_f32_bf16_v4(const float4* __restrict__ src,
                                __bf16* __restrict__ dst, int n4) {
  int i = blockIdx.x * blockDim.x + threadIdx.x;
  if (i < n4) {
    float4 f = src[i];
    v4bf o = { (__bf16)f.x, (__bf16)f.y, (__bf16)f.z, (__bf16)f.w };
    *(v4bf*)(dst + (size_t)i * 4) = o;
  }
}

// ---------------------------------------------------------------------------
// Ping-pong pipelined 32x64 wave-tile GEMM body.
// acc[s][c]: s = M sub-tile (16 rows), c = N sub-tile (16 cols).
// ---------------------------------------------------------------------------
DI void gemm32x64(const __bf16* arow0, const __bf16* arow1,
                  const __bf16* const bcol[4], int lane, v8f acc[2][4]) {
  const int aoff = (lane & 16) ? 8 : 0;
  const int boff = (lane & 16) ? 8 : 0;
  const __bf16* pa0 = arow0 + aoff;
  const __bf16* pa1 = arow1 + aoff;
  const __bf16* pb0 = bcol[0] + boff;
  const __bf16* pb1 = bcol[1] + boff;
  const __bf16* pb2 = bcol[2] + boff;
  const __bf16* pb3 = bcol[3] + boff;

  v16bf a0[2], a1[2], b0[2], b1[2], b2[2], b3[2];
  a0[0] = load_a_at(pa0);  a1[0] = load_a_at(pa1);
  b0[0] = load_b_at(pb0);  b1[0] = load_b_at(pb1);
  b2[0] = load_b_at(pb2);  b3[0] = load_b_at(pb3);

#pragma unroll 1
  for (int kk = 0; kk < NK - 2; kk += 2) {
    // prefetch k+1 into buffer 1
    a0[1] = load_a_at(pa0 + 32);  a1[1] = load_a_at(pa1 + 32);
    b0[1] = load_b_at(pb0 + 32);  b1[1] = load_b_at(pb1 + 32);
    b2[1] = load_b_at(pb2 + 32);  b3[1] = load_b_at(pb3 + 32);
    // compute k with buffer 0
    acc[0][0] = wmma_bf16(a0[0], b0[0], acc[0][0]);
    acc[0][1] = wmma_bf16(a0[0], b1[0], acc[0][1]);
    acc[0][2] = wmma_bf16(a0[0], b2[0], acc[0][2]);
    acc[0][3] = wmma_bf16(a0[0], b3[0], acc[0][3]);
    acc[1][0] = wmma_bf16(a1[0], b0[0], acc[1][0]);
    acc[1][1] = wmma_bf16(a1[0], b1[0], acc[1][1]);
    acc[1][2] = wmma_bf16(a1[0], b2[0], acc[1][2]);
    acc[1][3] = wmma_bf16(a1[0], b3[0], acc[1][3]);
    // prefetch k+2 into buffer 0
    a0[0] = load_a_at(pa0 + 64);  a1[0] = load_a_at(pa1 + 64);
    b0[0] = load_b_at(pb0 + 64);  b1[0] = load_b_at(pb1 + 64);
    b2[0] = load_b_at(pb2 + 64);  b3[0] = load_b_at(pb3 + 64);
    // compute k+1 with buffer 1
    acc[0][0] = wmma_bf16(a0[1], b0[1], acc[0][0]);
    acc[0][1] = wmma_bf16(a0[1], b1[1], acc[0][1]);
    acc[0][2] = wmma_bf16(a0[1], b2[1], acc[0][2]);
    acc[0][3] = wmma_bf16(a0[1], b3[1], acc[0][3]);
    acc[1][0] = wmma_bf16(a1[1], b0[1], acc[1][0]);
    acc[1][1] = wmma_bf16(a1[1], b1[1], acc[1][1]);
    acc[1][2] = wmma_bf16(a1[1], b2[1], acc[1][2]);
    acc[1][3] = wmma_bf16(a1[1], b3[1], acc[1][3]);
    pa0 += 64; pa1 += 64;
    pb0 += 64; pb1 += 64; pb2 += 64; pb3 += 64;
  }
  // peeled final pair (k = NK-2 in buffer 0; load NK-1 into buffer 1)
  a0[1] = load_a_at(pa0 + 32);  a1[1] = load_a_at(pa1 + 32);
  b0[1] = load_b_at(pb0 + 32);  b1[1] = load_b_at(pb1 + 32);
  b2[1] = load_b_at(pb2 + 32);  b3[1] = load_b_at(pb3 + 32);
  acc[0][0] = wmma_bf16(a0[0], b0[0], acc[0][0]);
  acc[0][1] = wmma_bf16(a0[0], b1[0], acc[0][1]);
  acc[0][2] = wmma_bf16(a0[0], b2[0], acc[0][2]);
  acc[0][3] = wmma_bf16(a0[0], b3[0], acc[0][3]);
  acc[1][0] = wmma_bf16(a1[0], b0[0], acc[1][0]);
  acc[1][1] = wmma_bf16(a1[0], b1[0], acc[1][1]);
  acc[1][2] = wmma_bf16(a1[0], b2[0], acc[1][2]);
  acc[1][3] = wmma_bf16(a1[0], b3[0], acc[1][3]);
  acc[0][0] = wmma_bf16(a0[1], b0[1], acc[0][0]);
  acc[0][1] = wmma_bf16(a0[1], b1[1], acc[0][1]);
  acc[0][2] = wmma_bf16(a0[1], b2[1], acc[0][2]);
  acc[0][3] = wmma_bf16(a0[1], b3[1], acc[0][3]);
  acc[1][0] = wmma_bf16(a1[1], b0[1], acc[1][0]);
  acc[1][1] = wmma_bf16(a1[1], b1[1], acc[1][1]);
  acc[1][2] = wmma_bf16(a1[1], b2[1], acc[1][2]);
  acc[1][3] = wmma_bf16(a1[1], b3[1], acc[1][3]);
}

// ---------------------------------------------------------------------------
// Kernel 1: Q/K/V projection GEMM.  y = x @ W^T + b  (M=8192, N=K=1024)
// blockIdx.z selects which projection; epilogue scatters to attention layouts:
//   Q,K: [b][h][t][256]   (h = n*4 + v/256, d = v%256)
//   V:   [b][n][h][d][t]  (h = v/128,       d = v%128)   (transposed for P*V)
// Workgroup = 8 waves -> 128x128 tile; wave = 32x64.
// ---------------------------------------------------------------------------
__global__ __launch_bounds__(256) void proj_qkv_kernel(
    const __bf16* __restrict__ xb,
    const __bf16* __restrict__ Wqb, const __bf16* __restrict__ Wkb,
    const __bf16* __restrict__ Wvb,
    const float* __restrict__ bq, const float* __restrict__ bk,
    const float* __restrict__ bv,
    __bf16* __restrict__ Qb, __bf16* __restrict__ Kb, __bf16* __restrict__ Vb)
{
  const int lane = threadIdx.x & 31;
  const int w    = threadIdx.x >> 5;
  const int wy   = w & 3;
  const int wx   = w >> 2;
  const int n0   = blockIdx.x * 128 + wx * 64;
  const int m0   = blockIdx.y * 128 + wy * 32;
  const int z    = blockIdx.z;

  const __bf16* Wb   = (z == 0) ? Wqb : (z == 1) ? Wkb : Wvb;
  const float*  bias = (z == 0) ? bq  : (z == 1) ? bk  : bv;

  const __bf16* arow0 = xb + (size_t)(m0 + (lane & 15)) * U;
  const __bf16* arow1 = arow0 + (size_t)16 * U;
  const __bf16* bcol[4];
#pragma unroll
  for (int c = 0; c < 4; ++c)
    bcol[c] = Wb + (size_t)(n0 + c * 16 + (lane & 15)) * U;

  v8f acc[2][4] = {};
  gemm32x64(arow0, arow1, bcol, lane, acc);

#pragma unroll
  for (int c = 0; c < 4; ++c) {
    const int v = n0 + c * 16 + (lane & 15);
    const float bval = bias[v];
#pragma unroll
    for (int s = 0; s < 2; ++s) {
#pragma unroll
      for (int r = 0; r < 8; ++r) {
        const int gm = m0 + s * 16 + r + ((lane & 16) ? 8 : 0);
        const float val = acc[s][c][r] + bval;
        const int b = gm >> 11;          // / (NCH*T)
        const int n = (gm >> 10) & 1;    // channel
        const int t = gm & (T - 1);
        if (z == 2) {
          const int h = v >> 7, d = v & 127;
          Vb[(((size_t)(b * NCH + n) * H + h) * DK + d) * T + t] = (__bf16)val;
        } else {
          const int h = n * 4 + (v >> 8), d = v & 255;
          __bf16* dst = (z == 0) ? Qb : Kb;
          dst[((size_t)(b * H + h) * T + t) * DCO + d] = (__bf16)val;
        }
      }
    }
  }
}

// ---------------------------------------------------------------------------
// Kernel 2: fused flash attention per (b,h).
// Workgroup = 8 waves = 128 query rows; wave owns a 16-row tile.
// Streams 64-key tiles; online softmax in log2 domain; ping-pong loads.
// grid = (T/128, H, B)
// ---------------------------------------------------------------------------
__global__ __launch_bounds__(256) void attn_kernel(
    const __bf16* __restrict__ Qb, const __bf16* __restrict__ Kb,
    const __bf16* __restrict__ Vb, __bf16* __restrict__ Ao)
{
  __shared__ __bf16 Plds[8][16][64];   // per-wave private P transpose buffer

  const int lane = threadIdx.x & 31;
  const int w    = threadIdx.x >> 5;
  const int t0   = blockIdx.x * 128;
  const int h    = blockIdx.y;
  const int b    = blockIdx.z;

  const int aoff = (lane & 16) ? 8 : 0;
  const int boff = (lane & 16) ? 8 : 0;

  const __bf16* Qbase = Qb + (size_t)(b * H + h) * T * DCO;
  const __bf16* Kbase = Kb + (size_t)(b * H + h) * T * DCO;
  const __bf16* qrow  =
      Qbase + (size_t)(t0 + w * 16 + (lane & 15)) * DCO + aoff;

  v8f o[16] = {};                      // 16 rows x 256 cols accumulator
  float m[8], l[8];
#pragma unroll
  for (int r = 0; r < 8; ++r) { m[r] = -__builtin_inff(); l[r] = 0.f; }

  const float kscale = 0.0625f * 1.44269504088896f;  // 1/sqrt(256) * log2(e)

  for (int s0 = 0; s0 < T; s0 += 64) {
    // ---- scores tile: 16 x 64 = Q(16x256) * K^T(256x64), ping-pong
    const __bf16* krow[4];
#pragma unroll
    for (int c = 0; c < 4; ++c)
      krow[c] = Kbase + (size_t)(s0 + c * 16 + (lane & 15)) * DCO + boff;

    v8f sc[4] = {};
    {
      v16bf qa[2], kb[2][4];
      qa[0] = load_a_at(qrow);
#pragma unroll
      for (int c = 0; c < 4; ++c) kb[0][c] = load_b_at(krow[c]);
#pragma unroll
      for (int kk = 0; kk < 8; kk += 2) {
        const int k1 = (kk + 1) * 32;
        qa[1] = load_a_at(qrow + k1);
#pragma unroll
        for (int c = 0; c < 4; ++c) kb[1][c] = load_b_at(krow[c] + k1);
#pragma unroll
        for (int c = 0; c < 4; ++c) sc[c] = wmma_bf16(qa[0], kb[0][c], sc[c]);
        const int k2 = (kk + 2 < 8) ? (kk + 2) * 32 : 0;   // const-folded
        qa[0] = load_a_at(qrow + k2);
#pragma unroll
        for (int c = 0; c < 4; ++c) kb[0][c] = load_b_at(krow[c] + k2);
#pragma unroll
        for (int c = 0; c < 4; ++c) sc[c] = wmma_bf16(qa[1], kb[1][c], sc[c]);
      }
    }

    // ---- prefetch next s-tile (K rows + V rows) to overlap softmax VALU
    if (s0 + 64 < T) {
#pragma unroll
      for (int c = 0; c < 4; ++c)
        __builtin_prefetch(
            Kbase + (size_t)(s0 + 64 + c * 16 + (lane & 15)) * DCO, 0, 3);
      const int dd = (lane & 15) * 16;   // touch a spread of V rows
      const int ch = dd >> 7, dk = dd & 127;
      __builtin_prefetch(
          Vb + (((size_t)(b * NCH + ch) * H + h) * DK + dk) * T + s0 + 64, 0, 3);
    }

    // ---- online softmax (log2 domain); row r lives in VGPR r, N across half
    float p[4][8];
#pragma unroll
    for (int r = 0; r < 8; ++r) {
      const float z0 = sc[0][r] * kscale, z1 = sc[1][r] * kscale;
      const float z2 = sc[2][r] * kscale, z3 = sc[3][r] * kscale;
      float rmax = fmaxf(fmaxf(z0, z1), fmaxf(z2, z3));
      rmax = fmaxf(rmax, __shfl_xor(rmax, 1, 32));
      rmax = fmaxf(rmax, __shfl_xor(rmax, 2, 32));
      rmax = fmaxf(rmax, __shfl_xor(rmax, 4, 32));
      rmax = fmaxf(rmax, __shfl_xor(rmax, 8, 32));
      const float mnew  = fmaxf(m[r], rmax);
      const float alpha = exp2f(m[r] - mnew);
      const float p0 = exp2f(z0 - mnew), p1 = exp2f(z1 - mnew);
      const float p2 = exp2f(z2 - mnew), p3 = exp2f(z3 - mnew);
      float rs = p0 + p1 + p2 + p3;
      rs += __shfl_xor(rs, 1, 32);
      rs += __shfl_xor(rs, 2, 32);
      rs += __shfl_xor(rs, 4, 32);
      rs += __shfl_xor(rs, 8, 32);
      l[r] = l[r] * alpha + rs;
      m[r] = mnew;
#pragma unroll
      for (int j = 0; j < 16; ++j) o[j][r] *= alpha;
      p[0][r] = p0; p[1][r] = p1; p[2][r] = p2; p[3][r] = p3;
    }

    // ---- transpose P (C layout -> A layout) through private LDS tile
    const int moff = (lane & 16) ? 8 : 0;
#pragma unroll
    for (int c = 0; c < 4; ++c)
#pragma unroll
      for (int r = 0; r < 8; ++r)
        Plds[w][r + moff][c * 16 + (lane & 15)] = (__bf16)p[c][r];
    asm volatile("s_wait_dscnt 0" ::: "memory");

    // ---- out += P(16x64) * V(64x256); V stored [b][n][h][d][t] (s-contig)
    const __bf16* prow = &Plds[w][lane & 15][0] + aoff;
    const v16bf pa0 = load_a_at(prow);
    const v16bf pa1 = load_a_at(prow + 32);

    const int dd0 = lane & 15;
    // V row base for output column dd = j*16 + dd0
    const __bf16* vrow[16];
#pragma unroll
    for (int j = 0; j < 16; ++j) {
      const int dd = j * 16 + dd0;
      const int ch = dd >> 7, dk = dd & 127;
      vrow[j] = Vb + (((size_t)(b * NCH + ch) * H + h) * DK + dk) * T
                + s0 + boff;
    }

    // flattened, ping-pong 32-step loop over (kk2, j), fully unrolled
    {
      v16bf vb[2];
      vb[0] = load_b_at(vrow[0]);
#pragma unroll
      for (int idx = 0; idx < 32; idx += 2) {
        const int j0 = idx & 15,        kk0 = idx >> 4;
        const int i1 = idx + 1;
        const int j1 = i1 & 15,         kk1 = i1 >> 4;
        vb[1] = load_b_at(vrow[j1] + kk1 * 32);
        o[j0] = wmma_bf16(kk0 ? pa1 : pa0, vb[0], o[j0]);
        const int i2 = (idx + 2 < 32) ? idx + 2 : 0;   // const-folded
        vb[0] = load_b_at(vrow[i2 & 15] + (i2 >> 4) * 32);
        o[j1] = wmma_bf16(kk1 ? pa1 : pa0, vb[1], o[j1]);
      }
    }
  }

  // ---- normalize and write Ao rows = (b, ch, t), cols u = h*128 + dk (bf16)
  float inv[8];
#pragma unroll
  for (int r = 0; r < 8; ++r) inv[r] = 1.0f / l[r];
#pragma unroll
  for (int j = 0; j < 16; ++j) {
    const int dd = j * 16 + (lane & 15);
    const int ch = dd >> 7, dk = dd & 127;
#pragma unroll
    for (int r = 0; r < 8; ++r) {
      const int t = t0 + w * 16 + r + ((lane & 16) ? 8 : 0);
      const size_t row = (size_t)(b * NCH + ch) * T + t;
      Ao[row * U + h * DK + dk] = (__bf16)(o[j][r] * inv[r]);
    }
  }
}

// ---------------------------------------------------------------------------
// Kernel 3: output projection. out = Ao @ Wo^T + bo, f32 result.
// ---------------------------------------------------------------------------
__global__ __launch_bounds__(256) void proj_o_kernel(
    const __bf16* __restrict__ Ao, const __bf16* __restrict__ Wob,
    const float* __restrict__ bo, float* __restrict__ out)
{
  const int lane = threadIdx.x & 31;
  const int w    = threadIdx.x >> 5;
  const int wy   = w & 3;
  const int wx   = w >> 2;
  const int n0   = blockIdx.x * 128 + wx * 64;
  const int m0   = blockIdx.y * 128 + wy * 32;

  const __bf16* arow0 = Ao + (size_t)(m0 + (lane & 15)) * U;
  const __bf16* arow1 = arow0 + (size_t)16 * U;
  const __bf16* bcol[4];
#pragma unroll
  for (int c = 0; c < 4; ++c)
    bcol[c] = Wob + (size_t)(n0 + c * 16 + (lane & 15)) * U;

  v8f acc[2][4] = {};
  gemm32x64(arow0, arow1, bcol, lane, acc);

#pragma unroll
  for (int c = 0; c < 4; ++c) {
    const int v = n0 + c * 16 + (lane & 15);
    const float bval = bo[v];
#pragma unroll
    for (int s = 0; s < 2; ++s) {
#pragma unroll
      for (int r = 0; r < 8; ++r) {
        const int gm = m0 + s * 16 + r + ((lane & 16) ? 8 : 0);
        out[(size_t)gm * U + v] = acc[s][c][r] + bval;
      }
    }
  }
}

// ---------------------------------------------------------------------------
extern "C" void kernel_launch(void* const* d_in, const int* in_sizes, int n_in,
                              void* d_out, int out_size, void* d_ws,
                              size_t ws_size, hipStream_t stream) {
  const float* x  = (const float*)d_in[0];
  const float* Wq = (const float*)d_in[2];
  const float* bq = (const float*)d_in[3];
  const float* Wk = (const float*)d_in[4];
  const float* bk = (const float*)d_in[5];
  const float* Wv = (const float*)d_in[6];
  const float* bv = (const float*)d_in[7];
  const float* Wo = (const float*)d_in[8];
  const float* bo = (const float*)d_in[9];
  float* out = (float*)d_out;

  char* ws = (char*)d_ws;
  __bf16* xb  = (__bf16*)(ws);                         // 16 MiB
  __bf16* Wqb = (__bf16*)(ws + (size_t)16 * 1048576);  //  2 MiB each
  __bf16* Wkb = (__bf16*)(ws + (size_t)18 * 1048576);
  __bf16* Wvb = (__bf16*)(ws + (size_t)20 * 1048576);
  __bf16* Wob = (__bf16*)(ws + (size_t)22 * 1048576);
  __bf16* Qb  = (__bf16*)(ws + (size_t)24 * 1048576);  // 16 MiB
  __bf16* Kb  = (__bf16*)(ws + (size_t)40 * 1048576);  // 16 MiB
  __bf16* Vb  = (__bf16*)(ws + (size_t)56 * 1048576);  // 16 MiB
  __bf16* Ao  = (__bf16*)(ws + (size_t)72 * 1048576);  // 16 MiB

  const int NX4 = MROWS * U / 4;   // 2M float4
  const int NW4 = U * U / 4;       // 256K float4
  cvt_f32_bf16_v4<<<(NX4 + 255) / 256, 256, 0, stream>>>((const float4*)x,  xb,  NX4);
  cvt_f32_bf16_v4<<<(NW4 + 255) / 256, 256, 0, stream>>>((const float4*)Wq, Wqb, NW4);
  cvt_f32_bf16_v4<<<(NW4 + 255) / 256, 256, 0, stream>>>((const float4*)Wk, Wkb, NW4);
  cvt_f32_bf16_v4<<<(NW4 + 255) / 256, 256, 0, stream>>>((const float4*)Wv, Wvb, NW4);
  cvt_f32_bf16_v4<<<(NW4 + 255) / 256, 256, 0, stream>>>((const float4*)Wo, Wob, NW4);

  proj_qkv_kernel<<<dim3(U / 128, MROWS / 128, 3), 256, 0, stream>>>(
      xb, Wqb, Wkb, Wvb, bq, bk, bv, Qb, Kb, Vb);

  attn_kernel<<<dim3(T / 128, H, BATCH), 256, 0, stream>>>(Qb, Kb, Vb, Ao);

  proj_o_kernel<<<dim3(U / 128, MROWS / 128), 256, 0, stream>>>(Ao, Wob, bo, out);
}
